// RAM_26405458936033
// MI455X (gfx1250) — compile-verified, matrix operand order checked by reading, not run on hardware
//
#include <hip/hip_runtime.h>
#include <math.h>

// ---- problem constants ----
constexpr int NB = 128;   // batch
constexpr int NS = 256;   // seq len
constexpr int NE = 300;   // embed dim
constexpr int NH = 300;   // hidden
constexpr int NA = 8;     // aspect len
constexpr int NC = 3;     // classes
constexpr int HP  = 320;            // padded E/H (mult of 32)
constexpr int KK2 = 2 * HP;         // GEMM K (x | h) = 640
constexpr int NG  = 4 * HP;         // padded gate dim = 1280
constexpr int MW  = 608;            // padded memory width (601 used)
constexpr int LDA = KK2 + 8;        // LDS A row stride (648 halves) -> bank-conflict free

typedef __attribute__((ext_vector_type(16))) _Float16 v16h;
typedef __attribute__((ext_vector_type(8)))  float    v8f;
typedef __attribute__((ext_vector_type(2)))  _Float16 h2;

union TileU { v16h v; h2 h[8]; };

__device__ __forceinline__ float sigm(float x) { return 1.f / (1.f + __expf(-x)); }

// Load a 16x32 f16 fragment per ISA 7.12.2 (16-bit A layout; B mirrored with
// lane<->column so reading the (N,K) row-major weight yields X @ W^T).
__device__ __forceinline__ v16h load_tile_f16(const _Float16* base, int ld) {
  const int lane = threadIdx.x & 31;
  const int r    = lane & 15;
  const int hi   = lane >> 4;
  TileU u;
  const h2* row = reinterpret_cast<const h2*>(base + (size_t)r * ld);
#pragma unroll
  for (int v = 0; v < 8; ++v) {
    int k0 = (v < 4) ? (2 * v + 8 * hi) : (16 + 2 * (v - 4) + 8 * hi);
    u.h[v] = row[k0 >> 1];
  }
  return u.v;
}

// ---------------- sequence lengths ----------------
__global__ void k_lens(const int* __restrict__ text, const int* __restrict__ aspect,
                       const int* __restrict__ leftc, int* __restrict__ lens) {
  int b = threadIdx.x;
  if (b >= NB) return;
  int ml = 0, ll = 0, al = 0;
  for (int s = 0; s < NS; ++s) ml += (text[b * NS + s] != 0);
  for (int s = 0; s < NS; ++s) ll += (leftc[b * NS + s] != 0);
  for (int a = 0; a < NA; ++a) al += (aspect[b * NA + a] != 0);
  lens[b] = ml; lens[NB + b] = ll; lens[2 * NB + b] = al;
}

// ------------- pack padded gate weights (f16) + combined bias -------------
__global__ void k_prep(const float* __restrict__ Wih_f, const float* __restrict__ Whh_f,
                       const float* __restrict__ bih_f, const float* __restrict__ bhh_f,
                       const float* __restrict__ Wih_b, const float* __restrict__ Whh_b,
                       const float* __restrict__ bih_b, const float* __restrict__ bhh_b,
                       _Float16* __restrict__ Wg, float* __restrict__ bg) {
  int g = blockIdx.x, d = blockIdx.y, k = threadIdx.x;   // g: gate row, k: K col
  const float* Wih = d ? Wih_b : Wih_f;
  const float* Whh = d ? Whh_b : Whh_f;
  int sec = g / HP, j = g % HP;
  float val = 0.f;
  if (j < NH) {
    if (k < HP) { if (k < NE) val = Wih[(size_t)(sec * NH + j) * NE + k]; }
    else        { int h = k - HP; if (h < NH) val = Whh[(size_t)(sec * NH + j) * NH + h]; }
  }
  Wg[((size_t)d * NG + g) * KK2 + k] = (_Float16)val;
  if (k == 0) {
    const float* bi = d ? bih_b : bih_f;
    const float* bh = d ? bhh_b : bhh_f;
    bg[d * NG + g] = (j < NH) ? (bi[sec * NH + j] + bh[sec * NH + j]) : 0.f;
  }
}

// ------------- embedding gather -> padded f16 x, fwd + reversed -------------
__global__ void k_embed(const int* __restrict__ text, const float* __restrict__ emb,
                        const int* __restrict__ lens, _Float16* __restrict__ xb) {
  int t = blockIdx.x, b = blockIdx.y, d = blockIdx.z, e = threadIdx.x;
  int tt = t;
  if (d == 1) { int ml = lens[b]; tt = (t < ml) ? (ml - 1 - t) : t; }
  int tok = text[b * NS + tt];
  float v = (e < NE) ? emb[(size_t)tok * NE + e] : 0.f;
  xb[(((size_t)d * NS + t) * NB + b) * HP + e] = (_Float16)v;
}

// ------------- zero h (both parities) and c state -------------
__global__ void k_init(_Float16* __restrict__ hs, float* __restrict__ cs) {
  int i = blockIdx.x * blockDim.x + threadIdx.x;
  if (i < 2 * 2 * NB * HP) hs[i] = (_Float16)0.f;
  if (i < 2 * NB * HP)     cs[i] = 0.f;
}

// ------------- one fused BiLSTM timestep: async LDS stage + WMMA + gates -----
// grid: (HP/16=20 hidden tiles, NB/32=4 batch tiles, 2 dirs), block: 128
// Waves = 4 gate sections; each wave does 2 m-subtiles -> 40 WMMAs.
__global__ __launch_bounds__(128)
void k_step(const _Float16* __restrict__ xbuf, _Float16* __restrict__ hstate,
            float* __restrict__ cstate, float* __restrict__ outbuf,
            const _Float16* __restrict__ Wg, const float* __restrict__ bg,
            const int* __restrict__ lens, int t) {
  __shared__ __align__(16) _Float16 shA[32][LDA];   // [32 rows][x(320)|h(320)] padded
  __shared__ float gbuf[4][32][17];
  const int d    = blockIdx.z;
  const int m0   = blockIdx.y * 32;   // batch tile base
  const int jb0  = blockIdx.x * 16;   // hidden tile base
  const int w    = threadIdx.x >> 5;  // gate section (i,f,g,o)
  const int lane = threadIdx.x & 31;
  const int r = lane & 15, hi = lane >> 4;
  const int par_r = t & 1, par_w = par_r ^ 1;
  const int tid = threadIdx.x;

  const _Float16* xbase = xbuf   + (((size_t)d * NS + t) * NB + m0) * HP;
  const _Float16* hbase = hstate + (((size_t)par_r * 2 + d) * NB + m0) * HP;

  // ---- async global->LDS stage of the 32x640 A tile (ASYNCcnt path) ----
  // 32 rows * 80 16B-chunks = 2560 chunks; 128 lanes x 20 iters.
#pragma unroll
  for (int it2 = 0; it2 < 20; ++it2) {
    int i  = it2 * 128 + tid;
    int rr = i / 80, cc = i % 80;
    const _Float16* src = (cc < 40) ? (xbase + (size_t)rr * HP + cc * 8)
                                    : (hbase + (size_t)rr * HP + (cc - 40) * 8);
    unsigned ldsoff = (unsigned)(uintptr_t)(void*)&shA[rr][cc * 8];
    asm volatile("global_load_async_to_lds_b128 %0, %1, off"
                 :: "v"(ldsoff), "v"(src) : "memory");
  }
  asm volatile("s_wait_asynccnt 0x0" ::: "memory");
  __syncthreads();

  const _Float16* wbase = Wg + ((size_t)d * NG + (w * HP + jb0)) * KK2;
  v8f acc0 = {}, acc1 = {};
#pragma unroll
  for (int kk = 0; kk < KK2 / 32; ++kk) {
    v16h bm = load_tile_f16(wbase + kk * 32, KK2);
    v16h a0 = load_tile_f16(&shA[0][0]  + kk * 32, LDA);
    v16h a1 = load_tile_f16(&shA[16][0] + kk * 32, LDA);
    acc0 = __builtin_amdgcn_wmma_f32_16x16x32_f16(false, a0, false, bm,
                                                  (short)0, acc0, false, false);
    acc1 = __builtin_amdgcn_wmma_f32_16x16x32_f16(false, a1, false, bm,
                                                  (short)0, acc1, false, false);
  }
  float bias = bg[(size_t)d * NG + w * HP + jb0 + r];
#pragma unroll
  for (int j = 0; j < 8; ++j) {
    gbuf[w][j + 8 * hi][r]      = acc0[j] + bias;
    gbuf[w][16 + j + 8 * hi][r] = acc1[j] + bias;
  }
  __syncthreads();

#pragma unroll
  for (int it = 0; it < 4; ++it) {
    int e = tid + it * 128;          // 512 (b,j) cells
    int m = e >> 4, j = e & 15;
    int b = m0 + m, jj = jb0 + j;
    float ig = sigm(gbuf[0][m][j]);
    float fg = sigm(gbuf[1][m][j]);
    float gg = tanhf(gbuf[2][m][j]);
    float og = sigm(gbuf[3][m][j]);
    size_t sidx = ((size_t)d * NB + b) * HP + jj;
    float c_old = cstate[sidx];
    float c_new = fg * c_old + ig * gg;
    float h_new = og * tanhf(c_new);
    bool  msk   = t < lens[b];
    float h_old = (float)hstate[(((size_t)par_r * 2 + d) * NB + b) * HP + jj];
    cstate[sidx] = msk ? c_new : c_old;
    hstate[(((size_t)par_w * 2 + d) * NB + b) * HP + jj] = (_Float16)(msk ? h_new : h_old);
    outbuf[(((size_t)d * NB + b) * NS + t) * HP + jj] = msk ? h_new : 0.f;
  }
}

// ------------- locationed memory: weight + un-reverse backward + position ----
__global__ void k_mem(const float* __restrict__ ob, const int* __restrict__ lens,
                      float* __restrict__ mem) {
  int t = blockIdx.x, b = blockIdx.y, dd = threadIdx.x;
  if (dd >= 2 * NH + 1) return;
  float ml = (float)lens[b], ll = (float)lens[NB + b], al = (float)lens[2 * NB + b];
  float pos = (float)t;
  bool in_left  = pos < ll;
  bool in_right = (pos >= ll + al) && (pos < ml);
  float w = in_left ? (1.f - (ll - pos) / ml)
                    : (in_right ? (1.f - (pos - ll - al + 1.f) / ml) : 1.f);
  float u = in_left ? (pos - ll) : (in_right ? (pos - ll - al + 1.f) : 0.f);
  float val;
  if (dd < NH) {
    val = w * ob[(((size_t)0 * NB + b) * NS + t) * HP + dd];
  } else if (dd < 2 * NH) {
    int mli = lens[b];
    int tt = (t < mli) ? (mli - 1 - t) : t;
    val = w * ob[(((size_t)1 * NB + b) * NS + tt) * HP + (dd - NH)];
  } else {
    val = u;
  }
  mem[((size_t)b * NS + t) * MW + dd] = val;
}

// ------------- aspect mean embedding + zero et -------------
__global__ void k_asp(const int* __restrict__ aspect, const float* __restrict__ emb,
                      const int* __restrict__ lens, float* __restrict__ asp,
                      float* __restrict__ etA) {
  int b = blockIdx.x, e = threadIdx.x;
  if (e >= NH) return;
  float s = 0.f;
  for (int a = 0; a < NA; ++a) s += emb[(size_t)aspect[b * NA + a] * NE + e];
  asp[b * NH + e] = s / (float)lens[2 * NB + b];
  etA[b * NH + e] = 0.f;
}

// ------------- attention scores + softmax over s -------------
__global__ __launch_bounds__(256)
void k_scores(const float* __restrict__ mem, const float* __restrict__ et,
              const float* __restrict__ asp, const float* __restrict__ attW,
              const float* __restrict__ attb, float* __restrict__ alpha) {
  __shared__ float red[256];
  int b = blockIdx.x, tid = threadIdx.x;
  float part = 0.f;
  for (int q = tid; q < 2 * NH; q += 256) {
    float v = (q < NH) ? et[b * NH + q] : asp[b * NH + (q - NH)];
    part += attW[2 * NH + 1 + q] * v;
  }
  red[tid] = part; __syncthreads();
  for (int st = 128; st > 0; st >>= 1) { if (tid < st) red[tid] += red[tid + st]; __syncthreads(); }
  float cdot = red[0] + attb[0];
  __syncthreads();
  const float* mrow = mem + ((size_t)b * NS + tid) * MW;
  float sc = cdot;
  for (int q = 0; q < 2 * NH + 1; ++q) sc += attW[q] * mrow[q];
  red[tid] = sc; __syncthreads();
  for (int st = 128; st > 0; st >>= 1) { if (tid < st) red[tid] = fmaxf(red[tid], red[tid + st]); __syncthreads(); }
  float mx = red[0]; __syncthreads();
  float ex = __expf(sc - mx);
  red[tid] = ex; __syncthreads();
  for (int st = 128; st > 0; st >>= 1) { if (tid < st) red[tid] += red[tid + st]; __syncthreads(); }
  alpha[(size_t)b * NS + tid] = ex / red[0];
}

// ------------- i = sum_s alpha * memory -------------
__global__ void k_isum(const float* __restrict__ mem, const float* __restrict__ alpha,
                       float* __restrict__ ivec) {
  int b = blockIdx.x, dd = threadIdx.x;
  if (dd >= 2 * NH + 1) return;
  float acc = 0.f;
  for (int s = 0; s < NS; ++s)
    acc += alpha[(size_t)b * NS + s] * mem[((size_t)b * NS + s) * MW + dd];
  ivec[b * MW + dd] = acc;
}

// ------------- GRU cell update -------------
__global__ void k_gru(const float* __restrict__ ivec, const float* __restrict__ et_in,
                      const float* __restrict__ gWih, const float* __restrict__ gWhh,
                      const float* __restrict__ gbih, const float* __restrict__ gbhh,
                      float* __restrict__ et_out) {
  int b = blockIdx.x, j = threadIdx.x;
  if (j >= NE) return;
  const int KI = 2 * NH + 1;  // 601
  float gi_r = gbih[j], gi_z = gbih[NE + j], gi_n = gbih[2 * NE + j];
  for (int q = 0; q < KI; ++q) {
    float iv = ivec[b * MW + q];
    gi_r += gWih[(size_t)j * KI + q] * iv;
    gi_z += gWih[(size_t)(NE + j) * KI + q] * iv;
    gi_n += gWih[(size_t)(2 * NE + j) * KI + q] * iv;
  }
  float gh_r = gbhh[j], gh_z = gbhh[NE + j], gh_n = gbhh[2 * NE + j];
  for (int q = 0; q < NE; ++q) {
    float hv = et_in[b * NE + q];
    gh_r += gWhh[(size_t)j * NE + q] * hv;
    gh_z += gWhh[(size_t)(NE + j) * NE + q] * hv;
    gh_n += gWhh[(size_t)(2 * NE + j) * NE + q] * hv;
  }
  float rg = sigm(gi_r + gh_r);
  float zg = sigm(gi_z + gh_z);
  float ng = tanhf(gi_n + rg * gh_n);
  et_out[b * NE + j] = (1.f - zg) * ng + zg * et_in[b * NE + j];
}

// ------------- dense head -------------
__global__ void k_dense(const float* __restrict__ et, const float* __restrict__ dW,
                        const float* __restrict__ db, float* __restrict__ out) {
  int tid = threadIdx.x;
  if (tid >= NB * NC) return;
  int b = tid / NC, c = tid % NC;
  float acc = db[c];
  for (int q = 0; q < NE; ++q) acc += et[b * NE + q] * dW[c * NE + q];
  out[b * NC + c] = acc;
}

extern "C" void kernel_launch(void* const* d_in, const int* in_sizes, int n_in,
                              void* d_out, int out_size, void* d_ws, size_t ws_size,
                              hipStream_t stream) {
  (void)in_sizes; (void)n_in; (void)out_size; (void)ws_size;
  const int*   text    = (const int*)d_in[0];
  const int*   aspect  = (const int*)d_in[1];
  const int*   leftc   = (const int*)d_in[2];
  const float* emb     = (const float*)d_in[3];
  const float* Wih_f   = (const float*)d_in[4];
  const float* Whh_f   = (const float*)d_in[5];
  const float* bih_f   = (const float*)d_in[6];
  const float* bhh_f   = (const float*)d_in[7];
  const float* Wih_b   = (const float*)d_in[8];
  const float* Whh_b   = (const float*)d_in[9];
  const float* bih_b   = (const float*)d_in[10];
  const float* bhh_b   = (const float*)d_in[11];
  const float* attW    = (const float*)d_in[12];
  const float* attb    = (const float*)d_in[13];
  const float* gWih    = (const float*)d_in[14];
  const float* gWhh    = (const float*)d_in[15];
  const float* gbih    = (const float*)d_in[16];
  const float* gbhh    = (const float*)d_in[17];
  const float* dW      = (const float*)d_in[18];
  const float* db      = (const float*)d_in[19];
  float* out = (float*)d_out;

  size_t off = 0;
  auto wsalloc = [&](size_t bytes) -> void* {
    void* p = (char*)d_ws + off;
    off += (bytes + 255) & ~(size_t)255;
    return p;
  };
  int*       lens  = (int*)      wsalloc((size_t)3 * NB * sizeof(int));
  _Float16*  Wg    = (_Float16*) wsalloc((size_t)2 * NG * KK2 * sizeof(_Float16));
  float*     bg    = (float*)    wsalloc((size_t)2 * NG * sizeof(float));
  _Float16*  xb    = (_Float16*) wsalloc((size_t)2 * NS * NB * HP * sizeof(_Float16));
  _Float16*  hs    = (_Float16*) wsalloc((size_t)2 * 2 * NB * HP * sizeof(_Float16));
  float*     cs    = (float*)    wsalloc((size_t)2 * NB * HP * sizeof(float));
  float*     ob    = (float*)    wsalloc((size_t)2 * NB * NS * HP * sizeof(float));
  float*     mem   = (float*)    wsalloc((size_t)NB * NS * MW * sizeof(float));
  float*     asp   = (float*)    wsalloc((size_t)NB * NH * sizeof(float));
  float*     etA   = (float*)    wsalloc((size_t)NB * NH * sizeof(float));
  float*     etB   = (float*)    wsalloc((size_t)NB * NH * sizeof(float));
  float*     alpha = (float*)    wsalloc((size_t)NB * NS * sizeof(float));
  float*     ivec  = (float*)    wsalloc((size_t)NB * MW * sizeof(float));

  k_lens<<<1, NB, 0, stream>>>(text, aspect, leftc, lens);
  k_prep<<<dim3(NG, 2), KK2, 0, stream>>>(Wih_f, Whh_f, bih_f, bhh_f,
                                          Wih_b, Whh_b, bih_b, bhh_b, Wg, bg);
  k_embed<<<dim3(NS, NB, 2), HP, 0, stream>>>(text, emb, lens, xb);
  k_init<<<(2 * 2 * NB * HP + 255) / 256, 256, 0, stream>>>(hs, cs);

  for (int t = 0; t < NS; ++t)
    k_step<<<dim3(HP / 16, NB / 32, 2), 128, 0, stream>>>(xb, hs, cs, ob, Wg, bg, lens, t);

  k_mem<<<dim3(NS, NB), MW, 0, stream>>>(ob, lens, mem);
  k_asp<<<NB, HP, 0, stream>>>(aspect, emb, lens, asp, etA);

  float* e_in = etA;
  float* e_out = etB;
  for (int h = 0; h < 3; ++h) {
    k_scores<<<NB, 256, 0, stream>>>(mem, e_in, asp, attW, attb, alpha);
    k_isum<<<NB, MW, 0, stream>>>(mem, alpha, ivec);
    k_gru<<<NB, HP, 0, stream>>>(ivec, e_in, gWih, gWhh, gbih, gbhh, e_out);
    float* tmp = e_in; e_in = e_out; e_out = tmp;
  }
  k_dense<<<1, NB * NC, 0, stream>>>(e_in, dW, db, out);
}
